// Attention_10350871183692
// MI455X (gfx1250) — matrix operand order repaired and checked by reading, not compiled, not run on hardware
//
#include <hip/hip_runtime.h>

// ---------------- problem constants ----------------
#define S_LEN 2048
#define DMODEL 4096
#define NHEADS 32
#define NKV 8
#define HEADD 128
#define NREP 4
#define SCALE 0.08838834764831845f   // 1/sqrt(128)

// ---------------- async global->LDS support (guarded) ----------------
#if defined(__has_builtin)
#  if __has_builtin(__builtin_amdgcn_global_load_async_to_lds_b128)
#    define USE_ASYNC_LDS 1
#  endif
#endif
#ifndef USE_ASYNC_LDS
#  define USE_ASYNC_LDS 0
#endif

#if USE_ASYNC_LDS
typedef int async_v4i __attribute__((vector_size(16)));
#define ASYNC_CP16(g, l, off)                                          \
    __builtin_amdgcn_global_load_async_to_lds_b128(                    \
        (__attribute__((address_space(1))) async_v4i*)(g),             \
        (__attribute__((address_space(3))) async_v4i*)(l), (off), 0)
#define ASYNC_CP64(g, l, B)                                            \
    do {                                                               \
        ASYNC_CP16(g, l, (B));      ASYNC_CP16(g, l, (B) + 16);        \
        ASYNC_CP16(g, l, (B) + 32); ASYNC_CP16(g, l, (B) + 48);        \
    } while (0)
#endif

__device__ __forceinline__ void wait_async0() {
#if USE_ASYNC_LDS
#if __has_builtin(__builtin_amdgcn_s_wait_asynccnt)
    __builtin_amdgcn_s_wait_asynccnt(0);
#else
    asm volatile("s_wait_asynccnt 0" ::: "memory");
#endif
#endif
}

// ---------------- bf16 / vector helpers ----------------
typedef __attribute__((ext_vector_type(2)))  __bf16 v2bf;
typedef __attribute__((ext_vector_type(8)))  __bf16 v8bf;
typedef __attribute__((ext_vector_type(16))) __bf16 v16bf;
typedef __attribute__((ext_vector_type(8)))  float  v8f;

__device__ __forceinline__ unsigned short f2bf(float f) {
    return __builtin_bit_cast(unsigned short, (__bf16)f);
}
__device__ __forceinline__ unsigned pk2(float a, float b) {
    v2bf t; t.x = (__bf16)a; t.y = (__bf16)b;
    return __builtin_bit_cast(unsigned, t);
}
__device__ __forceinline__ v8bf ld8(const unsigned short* p) {
    return *(const v8bf*)p;
}
__device__ __forceinline__ v16bf ld16(const unsigned short* p) {
    return *(const v16bf*)p;
}
__device__ __forceinline__ v16bf mk16(v8bf lo, v8bf hi) {
    union { v16bf v; v8bf h[2]; } u; u.h[0] = lo; u.h[1] = hi; return u.v;
}

// =====================================================================
// Elementwise f32 -> bf16 (4 per thread, vectorized)
// =====================================================================
__global__ __launch_bounds__(256) void cvt_bf16(
    const float* __restrict__ in, unsigned short* __restrict__ out) {
    size_t i = (size_t)blockIdx.x * 256 + threadIdx.x;
    float4 f = ((const float4*)in)[i];
    uint2 u; u.x = pk2(f.x, f.y); u.y = pk2(f.z, f.w);
    ((uint2*)out)[i] = u;
}

// =====================================================================
// GEMM:  C[M,N] = A[M,K] * W[N,K]^T   (bf16 in, f32 out)
// Block 256 threads = 8 waves; tile 128x128; K-tile 64 (2 WMMA k-steps).
// Double-buffered LDS: async copy of tile i+1 overlaps WMMA on tile i.
// =====================================================================
__global__ __launch_bounds__(256) void gemm_bf16_abt(
    const unsigned short* __restrict__ A, const unsigned short* __restrict__ W,
    float* __restrict__ C, int M, int N, int Kdim) {

    __shared__ unsigned short ldsA[2][128 * 64];   // 2 x 16 KB
    __shared__ unsigned short ldsW[2][128 * 64];   // 2 x 16 KB

    const int t     = threadIdx.x;
    const int lane  = t & 31;
    const int wave  = t >> 5;
    const int hs    = (lane >> 4) & 1;
    const int ln16  = lane & 15;
    const int waveM = wave >> 2;                // 0..1
    const int waveN = wave & 3;                 // 0..3

    const int mBase = blockIdx.y * 128;
    const int nBase = blockIdx.x * 128;

    v8f acc[4][2];
#pragma unroll
    for (int i = 0; i < 4; ++i)
#pragma unroll
        for (int j = 0; j < 2; ++j) acc[i][j] = (v8f)0.0f;

    const int srow = t >> 1;            // 0..127
    const int scb  = (t & 1) << 5;      // 0 or 32 (elements)

    auto stage = [&](int kb, int buf) {
        const unsigned short* sa = A + (size_t)(mBase + srow) * Kdim + kb + scb;
        const unsigned short* sw = W + (size_t)(nBase + srow) * Kdim + kb + scb;
        unsigned short* da = &ldsA[buf][srow * 64 + scb];
        unsigned short* dw = &ldsW[buf][srow * 64 + scb];
#if USE_ASYNC_LDS
        ASYNC_CP64(sa, da, 0);
        ASYNC_CP64(sw, dw, 0);
#else
#pragma unroll
        for (int j = 0; j < 4; ++j) ((uint4*)da)[j] = ((const uint4*)sa)[j];
#pragma unroll
        for (int j = 0; j < 4; ++j) ((uint4*)dw)[j] = ((const uint4*)sw)[j];
#endif
    };

    stage(0, 0);
    int buf = 0;
    for (int kb = 0; kb < Kdim; kb += 64) {
        wait_async0();
        __syncthreads();
        if (kb + 64 < Kdim) stage(kb + 64, buf ^ 1);  // overlap with WMMA below

#pragma unroll
        for (int ks = 0; ks < 2; ++ks) {
            const int kOff = 32 * ks;
            v16bf aF[4];
#pragma unroll
            for (int mi = 0; mi < 4; ++mi) {
                int row = waveM * 64 + mi * 16 + ln16;
                aF[mi] = ld16(&ldsA[buf][row * 64 + kOff + 16 * hs]);
            }
            v16bf bF[2];
#pragma unroll
            for (int ni = 0; ni < 2; ++ni) {
                int n = waveN * 32 + ni * 16 + ln16;
                bF[ni] = ld16(&ldsW[buf][n * 64 + kOff + 16 * hs]);
            }
#pragma unroll
            for (int mi = 0; mi < 4; ++mi)
#pragma unroll
                for (int ni = 0; ni < 2; ++ni)
                    acc[mi][ni] = __builtin_amdgcn_wmma_f32_16x16x32_bf16(
                        false, aF[mi], false, bF[ni], (short)0, acc[mi][ni],
                        false, false);
        }
        buf ^= 1;
    }

#pragma unroll
    for (int mi = 0; mi < 4; ++mi) {
#pragma unroll
        for (int ni = 0; ni < 2; ++ni) {
            int rowG = mBase + waveM * 64 + mi * 16 + 8 * hs;
            int colG = nBase + waveN * 32 + ni * 16 + ln16;
            float* cp = C + (size_t)rowG * N + colG;
#pragma unroll
            for (int r = 0; r < 8; ++r)
                cp[(size_t)r * N] = acc[mi][ni][r];
        }
    }
}

// =====================================================================
// RoPE (interleaved pairs): f32 in [S, nH, 128] -> bf16 out.
// headMajor=0: out [S, nH, 128]; headMajor=1: out [nH, S, 128].
// =====================================================================
__global__ __launch_bounds__(256) void rope_pack(
    const float* __restrict__ in, unsigned short* __restrict__ out,
    const float* __restrict__ cosb, const float* __restrict__ sinb,
    int nHeads, int headMajor) {
    int idx = blockIdx.x * blockDim.x + threadIdx.x;
    int total = S_LEN * nHeads * 64;
    if (idx >= total) return;
    int i = idx & 63;
    int h = (idx >> 6) % nHeads;
    int s = idx / (64 * nHeads);
    float c  = cosb[s * 64 + i];
    float sn = sinb[s * 64 + i];
    const float* p = in + ((size_t)s * nHeads + h) * HEADD + 2 * i;
    float tr = p[0], ti = p[1];
    float a = tr * c - ti * sn;
    float b = tr * sn + ti * c;
    size_t o = headMajor ? (((size_t)h * S_LEN + s) * HEADD + 2 * i)
                         : (((size_t)s * nHeads + h) * HEADD + 2 * i);
    *(unsigned*)(out + o) = pk2(a, b);
}

// =====================================================================
// V transpose: f32 [S, KV, 128] -> bf16 [KV, 128, S]  (coalesced stores)
// =====================================================================
__global__ __launch_bounds__(256) void transpose_v(
    const float* __restrict__ in, unsigned short* __restrict__ out) {
    int idx = blockIdx.x * blockDim.x + threadIdx.x;   // = (kv*128+d)*S + s
    int s = idx & (S_LEN - 1);
    int rest = idx >> 11;
    int d = rest & (HEADD - 1);
    int kv = rest >> 7;
    out[idx] = f2bf(in[((size_t)s * NKV + kv) * HEADD + d]);
}

// =====================================================================
// Causal flash attention, all-bf16 operands, double-buffered K/V tiles.
// Grid (S/64, H); 128 threads = 4 waves; wave owns 16 q rows.
// q: bf16 [S, H, 128];  k: bf16 [KV, S, 128];  vt: bf16 [KV, 128, S].
// =====================================================================
__global__ __launch_bounds__(128) void flash_attn(
    const unsigned short* __restrict__ qb, const unsigned short* __restrict__ kbuf,
    const unsigned short* __restrict__ vtb, float* __restrict__ o) {

    __shared__ unsigned short Ksh[2][64 * 128];  // [key][hd]   2 x 16 KB
    __shared__ unsigned short Vt[2][128 * 64];   // [hd][key]   2 x 16 KB
    __shared__ unsigned short Psh[4 * 16 * 64];  // per-wave P   8 KB

    const int t    = threadIdx.x;
    const int lane = t & 31;
    const int wave = t >> 5;
    const int hs   = (lane >> 4) & 1;
    const int ln16 = lane & 15;

    const int qRowBase = blockIdx.x * 64;
    const int head     = blockIdx.y;
    const int kvh      = head >> 2;             // NREP = 4

    const int key  = t & 63;
    const int half = t >> 6;

    auto stageKV = [&](int kbBase, int buf) {
        const unsigned short* ksrc =
            kbuf + ((size_t)kvh * S_LEN + kbBase + key) * HEADD + half * 64;
        unsigned short* kdst = &Ksh[buf][key * 128 + half * 64];
        const unsigned short* vsrc =
            vtb + ((size_t)kvh * HEADD + t) * S_LEN + kbBase;
        unsigned short* vdst = &Vt[buf][t * 64];
#if USE_ASYNC_LDS
        ASYNC_CP64(ksrc, kdst, 0);
        ASYNC_CP64(ksrc, kdst, 64);
        ASYNC_CP64(vsrc, vdst, 0);
        ASYNC_CP64(vsrc, vdst, 64);
#else
#pragma unroll
        for (int j = 0; j < 8; ++j)
            ((uint4*)kdst)[j] = ((const uint4*)ksrc)[j];
#pragma unroll
        for (int j = 0; j < 8; ++j)
            ((uint4*)vdst)[j] = ((const uint4*)vsrc)[j];
#endif
    };

    // ---- Q fragments (bf16 direct from global) ----
    stageKV(0, 0);
    const int qRow = qRowBase + wave * 16 + ln16;
    const unsigned short* qp = qb + ((size_t)qRow * NHEADS + head) * HEADD;
    v16bf qF[4];
#pragma unroll
    for (int kk = 0; kk < 4; ++kk) {
        qF[kk] = ld16(qp + 32 * kk + 16 * hs);
    }

    v8f oAcc[8];
#pragma unroll
    for (int i = 0; i < 8; ++i) oAcc[i] = (v8f)0.0f;
    float m[8], l[8];
#pragma unroll
    for (int r = 0; r < 8; ++r) { m[r] = -1e30f; l[r] = 0.0f; }

    unsigned short* pw = &Psh[wave * 16 * 64];

    int buf = 0;
    for (int kbBase = 0; kbBase <= qRowBase; kbBase += 64) {
        wait_async0();
        __syncthreads();
        if (kbBase + 64 <= qRowBase) stageKV(kbBase + 64, buf ^ 1); // overlap

        // ---- scores: S(16x64) = Q(16x128) . K^T ----
        v8f sAcc[4];
#pragma unroll
        for (int nt = 0; nt < 4; ++nt) sAcc[nt] = (v8f)0.0f;
#pragma unroll
        for (int kk = 0; kk < 4; ++kk) {
#pragma unroll
            for (int nt = 0; nt < 4; ++nt) {
                int n = nt * 16 + ln16;
                v16bf bF = ld16(&Ksh[buf][n * 128 + 32 * kk + 16 * hs]);
                sAcc[nt] = __builtin_amdgcn_wmma_f32_16x16x32_bf16(
                    false, qF[kk], false, bF, (short)0, sAcc[nt], false, false);
            }
        }

        // ---- online softmax (row = r + 8*hs, col = lane&15) ----
#pragma unroll
        for (int r = 0; r < 8; ++r) {
            int rowg = qRowBase + wave * 16 + r + 8 * hs;
            float mx = -1e30f;
#pragma unroll
            for (int nt = 0; nt < 4; ++nt) {
                float sc = sAcc[nt][r] * SCALE;
                int col = kbBase + nt * 16 + ln16;
                if (col > rowg) sc = -1e30f;
                sAcc[nt][r] = sc;
                mx = fmaxf(mx, sc);
            }
#pragma unroll
            for (int d = 1; d < 16; d <<= 1)
                mx = fmaxf(mx, __shfl_xor(mx, d, 32));
            float mNew = fmaxf(m[r], mx);
            float corr = __expf(m[r] - mNew);
            m[r] = mNew;
            float rs = 0.0f;
#pragma unroll
            for (int nt = 0; nt < 4; ++nt) {
                float p = __expf(sAcc[nt][r] - mNew);
                sAcc[nt][r] = p;
                rs += p;
            }
#pragma unroll
            for (int d = 1; d < 16; d <<= 1)
                rs += __shfl_xor(rs, d, 32);
            l[r] = l[r] * corr + rs;
#pragma unroll
            for (int ot = 0; ot < 8; ++ot) oAcc[ot][r] *= corr;
        }

        // ---- P -> LDS (bf16); per-wave region, dscnt-ordered ----
#pragma unroll
        for (int nt = 0; nt < 4; ++nt)
#pragma unroll
            for (int r = 0; r < 8; ++r)
                pw[(r + 8 * hs) * 64 + nt * 16 + ln16] = f2bf(sAcc[nt][r]);
        asm volatile("s_wait_dscnt 0" ::: "memory");

        // ---- O(16x128) += P(16x64) . V(64x128) ----
#pragma unroll
        for (int ks = 0; ks < 2; ++ks) {
            v16bf aF = ld16(&pw[ln16 * 64 + 32 * ks + 16 * hs]);
#pragma unroll
            for (int ot = 0; ot < 8; ++ot) {
                int n = ot * 16 + ln16;
                v16bf bF = ld16(&Vt[buf][n * 64 + 32 * ks + 16 * hs]);
                oAcc[ot] = __builtin_amdgcn_wmma_f32_16x16x32_bf16(
                    false, aF, false, bF, (short)0, oAcc[ot], false, false);
            }
        }
        buf ^= 1;
    }

    // ---- normalize and write O to [s][h*128 + hd] (f32) ----
    float inv[8];
#pragma unroll
    for (int r = 0; r < 8; ++r) inv[r] = 1.0f / l[r];
#pragma unroll
    for (int ot = 0; ot < 8; ++ot) {
#pragma unroll
        for (int r = 0; r < 8; ++r) {
            int rowg = qRowBase + wave * 16 + r + 8 * hs;
            int colg = head * HEADD + ot * 16 + ln16;
            o[(size_t)rowg * DMODEL + colg] = oAcc[ot][r] * inv[r];
        }
    }
}

// =====================================================================
// Launch
// =====================================================================
extern "C" void kernel_launch(void* const* d_in, const int* in_sizes, int n_in,
                              void* d_out, int out_size, void* d_ws, size_t ws_size,
                              hipStream_t stream) {
    const float* x    = (const float*)d_in[0];
    const float* fcos = (const float*)d_in[1];
    const float* fsin = (const float*)d_in[2];
    const float* wq   = (const float*)d_in[3];
    const float* wk   = (const float*)d_in[4];
    const float* wv   = (const float*)d_in[5];
    const float* wo   = (const float*)d_in[6];
    float* out = (float*)d_out;

    // ---- workspace slots (lifetime-reused), ~134 MB total ----
    char* p = (char*)d_ws;
    unsigned short* xb   = (unsigned short*)p; p += (size_t)8388608  * 2; // x bf16
    unsigned short* wqb  = (unsigned short*)p; p += (size_t)16777216 * 2; // wq / wo bf16
    unsigned short* wkb  = (unsigned short*)p; p += (size_t)4194304  * 2; // wk bf16 / k bf16
    unsigned short* wvb  = (unsigned short*)p; p += (size_t)4194304  * 2; // wv bf16 / vt bf16
    float*          qf32 = (float*)p;          p += (size_t)8388608  * 4; // q f32 / attn f32
    float*          kf32 = (float*)p;          p += (size_t)2097152  * 4;
    float*          vf32 = (float*)p;          p += (size_t)2097152  * 4;
    unsigned short* qb16 = (unsigned short*)p; p += (size_t)8388608  * 2;
    unsigned short* wob   = wqb;
    unsigned short* kb16  = wkb;
    unsigned short* vtb16 = wvb;
    float*          abuf  = qf32;
    unsigned short* ab16  = xb;

    // ---- one-time f32 -> bf16 conversions ----
    cvt_bf16<<<8388608 / 1024, 256, 0, stream>>>(x, xb);
    cvt_bf16<<<16777216 / 1024, 256, 0, stream>>>(wq, wqb);

    // ---- projections (bf16 WMMA) ----
    gemm_bf16_abt<<<dim3(DMODEL / 128, S_LEN / 128), 256, 0, stream>>>(
        xb, wqb, qf32, S_LEN, DMODEL, DMODEL);
    cvt_bf16<<<4194304 / 1024, 256, 0, stream>>>(wk, wkb);
    gemm_bf16_abt<<<dim3((NKV * HEADD) / 128, S_LEN / 128), 256, 0, stream>>>(
        xb, wkb, kf32, S_LEN, NKV * HEADD, DMODEL);
    cvt_bf16<<<4194304 / 1024, 256, 0, stream>>>(wv, wvb);
    gemm_bf16_abt<<<dim3((NKV * HEADD) / 128, S_LEN / 128), 256, 0, stream>>>(
        xb, wvb, vf32, S_LEN, NKV * HEADD, DMODEL);
    cvt_bf16<<<16777216 / 1024, 256, 0, stream>>>(wo, wob);   // reuses wqb slot

    // ---- RoPE + pack to bf16 (q: [S,H,128], k: [KV,S,128]) ----
    rope_pack<<<(S_LEN * NHEADS * 64) / 256, 256, 0, stream>>>(
        qf32, qb16, fcos, fsin, NHEADS, 0);
    rope_pack<<<(S_LEN * NKV * 64) / 256, 256, 0, stream>>>(
        kf32, kb16, fcos, fsin, NKV, 1);
    // ---- V -> bf16 transposed [KV,128,S] ----
    transpose_v<<<(S_LEN * NKV * HEADD) / 256, 256, 0, stream>>>(vf32, vtb16);

    // ---- flash attention ----
    flash_attn<<<dim3(S_LEN / 64, NHEADS), 128, 0, stream>>>(
        qb16, kb16, vtb16, abuf);

    // ---- output projection ----
    cvt_bf16<<<8388608 / 1024, 256, 0, stream>>>(abuf, ab16);
    gemm_bf16_abt<<<dim3(DMODEL / 128, S_LEN / 128), 256, 0, stream>>>(
        ab16, wob, out, S_LEN, DMODEL, DMODEL);
}
